// Model_16423954940206
// MI455X (gfx1250) — matrix operand order repaired and checked by reading, not compiled, not run on hardware
//
#include <hip/hip_runtime.h>
#include <hip/hip_bf16.h>

// ---------------------------------------------------------------------------
// SRU (2-layer, bidirectional) for MI455X / gfx1250.
// GEMMs: bf16 v_wmma_f32_16x16x32_bf16, double-buffered LDS filled by
// global_load_async_to_lds_b128 (ASYNCcnt pipeline). Scans: 16384 parallel
// recurrence chains, coalesced HBM streaming.
// ---------------------------------------------------------------------------

typedef __bf16         v16bf __attribute__((ext_vector_type(16)));
typedef float          v8f   __attribute__((ext_vector_type(8)));
typedef unsigned int   v4u   __attribute__((ext_vector_type(4)));

#define SRU_N 1024
#define SRU_L 1024
#define SRU_B 8

// round-to-nearest-even f32 -> bf16 (bit pattern)
__device__ __forceinline__ unsigned short f2bf(float x) {
    unsigned int u = __float_as_uint(x);
    u += 0x7FFFu + ((u >> 16) & 1u);
    return (unsigned short)(u >> 16);
}
__device__ __forceinline__ float bf2f(unsigned short h) {
    return __uint_as_float(((unsigned int)h) << 16);
}
__device__ __forceinline__ float sigmoidf(float x) {
    return 1.0f / (1.0f + __expf(-x));
}

// ---- gfx1250 async global->LDS copy (16B per lane), ASYNCcnt tracked ------
__device__ __forceinline__ void async_b128(unsigned int lds_addr, const void* gaddr) {
    asm volatile("global_load_async_to_lds_b128 %0, %1, off"
                 :: "v"(lds_addr), "v"(gaddr)
                 : "memory");
}
__device__ __forceinline__ void wait_async0() {
    asm volatile("s_wait_asynccnt 0x0" ::: "memory");
}
// generic (flat) pointer to LDS byte offset: hw truncates to addr[31:0]
__device__ __forceinline__ unsigned int lds_off(const void* p) {
    return (unsigned int)(unsigned long long)p;
}

// ---------------------------------------------------------------------------
// Conversion kernels
// ---------------------------------------------------------------------------

// emb (B, L, N) f32  ->  xbf (L*B, N) bf16   (transpose to time-major)
__global__ __launch_bounds__(256) void cvt_x_kernel(const float* __restrict__ emb,
                                                    unsigned short* __restrict__ xbf) {
    long gid = (long)blockIdx.x * blockDim.x + threadIdx.x;   // (l*B + b)*N + j
    int j  = (int)(gid % SRU_N);
    long r = gid / SRU_N;
    int b  = (int)(r % SRU_B);
    int l  = (int)(r / SRU_B);
    xbf[gid] = f2bf(emb[((long)b * SRU_L + l) * SRU_N + j]);
}

// w (K, Nout) f32 -> wt (Nout, K) bf16  (transpose so B fragments are K-contig)
__global__ __launch_bounds__(256) void cvt_wT_kernel(const float* __restrict__ w,
                                                     unsigned short* __restrict__ wt,
                                                     int K, int Nout) {
    long gid = (long)blockIdx.x * blockDim.x + threadIdx.x;   // n*K + k
    if (gid >= (long)K * Nout) return;
    int k = (int)(gid % K);
    int n = (int)(gid / K);
    wt[gid] = f2bf(w[(long)k * Nout + n]);
}

// ---------------------------------------------------------------------------
// bf16 WMMA GEMM:  C(M,Nn) = A(M,K) * Bt(Nn,K)^T   (A, Bt row-major bf16)
// Block tile 128x128, BK=32, 256 threads = 8 waves (4 in M x 2 in N),
// each wave owns 32x64 (2x4 WMMA tiles). LDS double-buffered, filled with
// global_load_async_to_lds_b128.
// ---------------------------------------------------------------------------
#define BM 128
#define BN 128
#define BK 32

__global__ __launch_bounds__(256) void gemm_bf16_wmma(
    const unsigned short* __restrict__ A,
    const unsigned short* __restrict__ Bt,
    float* __restrict__ C,
    int M, int Nn, int K)
{
    __shared__ __align__(16) unsigned short As[2][BM][BK];
    __shared__ __align__(16) unsigned short Bs[2][BN][BK];

    const int tid   = threadIdx.x;
    const int lane  = tid & 31;
    const int wave  = tid >> 5;           // 0..7
    const int wm    = (wave & 3) * 32;    // wave M offset inside tile
    const int wn    = (wave >> 2) * 64;   // wave N offset inside tile
    const int l16   = lane & 15;
    const int lhalf = lane >> 4;          // 0 / 1

    const int bm = blockIdx.x * BM;
    const int bn = blockIdx.y * BN;

    // async staging indexing: 16B per lane; tile (128x32 bf16 = 8KB) needs
    // two issues of 256 lanes x 16B. issue i covers rows [i*64, i*64+64).
    const int srow = tid >> 2;            // 0..63
    const int scol = (tid & 3) * 8;       // 0,8,16,24

    v8f acc[2][4] = {};

    const int iters = K / BK;

    // ---- prologue: stage k-slab 0 into buffer 0 ----
    {
        const unsigned short* ga = &A[((long)(bm + srow)) * K + scol];
        const unsigned short* gb = &Bt[((long)(bn + srow)) * K + scol];
        async_b128(lds_off(&As[0][srow][scol]), ga);
        async_b128(lds_off(&As[0][srow + 64][scol]), ga + (long)64 * K);
        async_b128(lds_off(&Bs[0][srow][scol]), gb);
        async_b128(lds_off(&Bs[0][srow + 64][scol]), gb + (long)64 * K);
    }

    for (int i = 0; i < iters; ++i) {
        const int cur = i & 1;

        wait_async0();        // this wave's async fills done
        __syncthreads();      // everyone's fills done; prev buffer free

        if (i + 1 < iters) {  // stage next k-slab into the other buffer
            const int kk = (i + 1) * BK;
            const int nxt = cur ^ 1;
            const unsigned short* ga = &A[((long)(bm + srow)) * K + kk + scol];
            const unsigned short* gb = &Bt[((long)(bn + srow)) * K + kk + scol];
            async_b128(lds_off(&As[nxt][srow][scol]), ga);
            async_b128(lds_off(&As[nxt][srow + 64][scol]), ga + (long)64 * K);
            async_b128(lds_off(&Bs[nxt][srow][scol]), gb);
            async_b128(lds_off(&Bs[nxt][srow + 64][scol]), gb + (long)64 * K);
        }

        // -------- WMMA fragments from LDS --------
        // A 16x32 bf16: lane holds row M=l16; K = {c..c+7, c+16..c+23}, c=lhalf*8
        v16bf afrag[2];
#pragma unroll
        for (int mt = 0; mt < 2; ++mt) {
            const unsigned short* p = &As[cur][wm + mt * 16 + l16][lhalf * 8];
            union { v4u q[2]; v16bf v; } u;
            u.q[0] = *(const v4u*)p;
            u.q[1] = *(const v4u*)(p + 16);
            afrag[mt] = u.v;
        }
        // B 32x16 bf16: lane holds col N=l16; K = {c..c+7, c+16..c+23}, c=lhalf*8
        v16bf bfrag[4];
#pragma unroll
        for (int nt = 0; nt < 4; ++nt) {
            const unsigned short* p = &Bs[cur][wn + nt * 16 + l16][lhalf * 8];
            union { v4u q[2]; v16bf v; } u;
            u.q[0] = *(const v4u*)p;
            u.q[1] = *(const v4u*)(p + 16);
            bfrag[nt] = u.v;
        }

#pragma unroll
        for (int mt = 0; mt < 2; ++mt)
#pragma unroll
            for (int nt = 0; nt < 4; ++nt)
                acc[mt][nt] = __builtin_amdgcn_wmma_f32_16x16x32_bf16(
                    /*neg_a=*/false, afrag[mt],
                    /*neg_b=*/false, bfrag[nt],
                    /*c_mod=*/(short)0, acc[mt][nt],
                    /*reuse_a=*/false, /*reuse_b=*/false);
    }

    // -------- epilogue: C/D layout VGPR r -> M = mt*16 + lhalf*8 + r --------
#pragma unroll
    for (int mt = 0; mt < 2; ++mt)
#pragma unroll
        for (int nt = 0; nt < 4; ++nt)
#pragma unroll
            for (int r = 0; r < 8; ++r) {
                int row = bm + wm + mt * 16 + lhalf * 8 + r;
                int col = bn + wn + nt * 16 + l16;
                C[(long)row * Nn + col] = acc[mt][nt][r];
            }
}

// ---------------------------------------------------------------------------
// Scan kernels: one thread per (b, dir, channel) recurrence chain.
// Layer 1: u (L,B,2,4,N), skip = u[...,3,:]. Layer 2: u (L,B,2,3,N), skip = h1.
// ---------------------------------------------------------------------------
__global__ __launch_bounds__(256) void scan1_kernel(
    const float* __restrict__ u,        // (L*B, 8N)
    const float* __restrict__ hidden,   // (2, B, 2N), layer 0 used
    const int*   __restrict__ mask,     // (B, L)
    const float* __restrict__ b1,       // (2, 2, N)
    unsigned short* __restrict__ h1,    // (L*B, 2N) bf16
    float* __restrict__ hid_out)        // (2, B, 2N), write layer 0
{
    int gid = blockIdx.x * blockDim.x + threadIdx.x;  // (b*2 + d)*N + j
    int j  = gid % SRU_N;
    int bd = gid / SRU_N;
    int d  = bd & 1;
    int b  = bd >> 1;

    float bfb = b1[(d * 2 + 0) * SRU_N + j];
    float brb = b1[(d * 2 + 1) * SRU_N + j];
    float c   = hidden[((long)0 * SRU_B + b) * 2 * SRU_N + d * SRU_N + j];

    int l = d ? (SRU_L - 1) : 0;
    const int step = d ? -1 : 1;
    for (int t = 0; t < SRU_L; ++t, l += step) {
        long row = (long)l * SRU_B + b;
        const float* up = u + (row * 8 + d * 4) * (long)SRU_N + j;
        float xt   = up[0];
        float f    = sigmoidf(up[SRU_N] + bfb);
        float r    = sigmoidf(up[2 * SRU_N] + brb);
        float skip = up[3 * SRU_N];
        bool  pad  = (mask[(long)b * SRU_L + l] == 0);

        float cn = f * c + (1.0f - f) * xt;
        cn = pad ? c : cn;
        float h = pad ? 0.0f : (r * cn + (1.0f - r) * skip);
        c = cn;
        h1[row * 2 * SRU_N + d * SRU_N + j] = f2bf(h);
    }
    hid_out[((long)0 * SRU_B + b) * 2 * SRU_N + d * SRU_N + j] = c;
}

__global__ __launch_bounds__(256) void scan2_kernel(
    const float* __restrict__ u,        // (L*B, 6N)
    const float* __restrict__ hidden,   // (2, B, 2N), layer 1 used
    const int*   __restrict__ mask,     // (B, L)
    const float* __restrict__ b2,       // (2, 2, N)
    const unsigned short* __restrict__ h1,  // (L*B, 2N) bf16 (skip input)
    float* __restrict__ outF,           // (L, B, N)
    float* __restrict__ outB,           // (L, B, N)
    float* __restrict__ hid_out)        // (2, B, 2N), write layer 1
{
    int gid = blockIdx.x * blockDim.x + threadIdx.x;
    int j  = gid % SRU_N;
    int bd = gid / SRU_N;
    int d  = bd & 1;
    int b  = bd >> 1;

    float bfb = b2[(d * 2 + 0) * SRU_N + j];
    float brb = b2[(d * 2 + 1) * SRU_N + j];
    float c   = hidden[((long)1 * SRU_B + b) * 2 * SRU_N + d * SRU_N + j];
    float* outp = d ? outB : outF;

    int l = d ? (SRU_L - 1) : 0;
    const int step = d ? -1 : 1;
    for (int t = 0; t < SRU_L; ++t, l += step) {
        long row = (long)l * SRU_B + b;
        const float* up = u + (row * 6 + d * 3) * (long)SRU_N + j;
        float xt   = up[0];
        float f    = sigmoidf(up[SRU_N] + bfb);
        float r    = sigmoidf(up[2 * SRU_N] + brb);
        float skip = bf2f(h1[row * 2 * SRU_N + d * SRU_N + j]);
        bool  pad  = (mask[(long)b * SRU_L + l] == 0);

        float cn = f * c + (1.0f - f) * xt;
        cn = pad ? c : cn;
        float h = pad ? 0.0f : (r * cn + (1.0f - r) * skip);
        c = cn;
        outp[row * SRU_N + j] = h;
    }
    hid_out[((long)1 * SRU_B + b) * 2 * SRU_N + d * SRU_N + j] = c;
}

// ---------------------------------------------------------------------------
// Launch
// ---------------------------------------------------------------------------
extern "C" void kernel_launch(void* const* d_in, const int* in_sizes, int n_in,
                              void* d_out, int out_size, void* d_ws, size_t ws_size,
                              hipStream_t stream) {
    const float* emb    = (const float*)d_in[0];   // (B, L, N)
    const float* hidden = (const float*)d_in[1];   // (2, B, 2N)
    const int*   mask   = (const int*)  d_in[2];   // (B, L)
    const float* w1     = (const float*)d_in[3];   // (N, 8N)
    const float* b1     = (const float*)d_in[4];   // (2, 2, N)
    const float* w2     = (const float*)d_in[5];   // (2N, 6N)
    const float* b2     = (const float*)d_in[6];   // (2, 2, N)

    const long LB  = (long)SRU_L * SRU_B;          // 8192
    const long LBN = LB * SRU_N;

    float* outF = (float*)d_out;                   // (L, B, N)
    float* outB = outF + LBN;                      // (L, B, N)
    float* hid  = outF + 2 * LBN;                  // (2, B, 2N)

    // workspace layout
    char* wsp = (char*)d_ws;
    unsigned short* xbf = (unsigned short*)wsp; wsp += LBN * 2;                 // 16 MB
    unsigned short* w1t = (unsigned short*)wsp; wsp += (long)8 * SRU_N * SRU_N * 2; // 16 MB (8N x N)
    unsigned short* w2t = (unsigned short*)wsp; wsp += (long)12 * SRU_N * SRU_N * 2;// 24 MB (6N x 2N)
    unsigned short* h1b = (unsigned short*)wsp; wsp += LB * 2 * SRU_N * 2;      // 32 MB
    float* u = (float*)wsp;                        // 256 MB, reused for u1 then u2

    // 1) convert activations and weights to bf16 (weights transposed to NxK)
    cvt_x_kernel<<<(unsigned)(LBN / 256), 256, 0, stream>>>(emb, xbf);
    cvt_wT_kernel<<<(unsigned)(((long)SRU_N * 8 * SRU_N) / 256), 256, 0, stream>>>(
        w1, w1t, SRU_N, 8 * SRU_N);
    cvt_wT_kernel<<<(unsigned)(((long)2 * SRU_N * 6 * SRU_N) / 256), 256, 0, stream>>>(
        w2, w2t, 2 * SRU_N, 6 * SRU_N);

    // 2) GEMM1: (8192 x 1024) x (1024 x 8192) -> u1
    {
        dim3 grid((unsigned)(LB / BM), (unsigned)(8 * SRU_N / BN));
        gemm_bf16_wmma<<<grid, 256, 0, stream>>>(xbf, w1t, u,
                                                 (int)LB, 8 * SRU_N, SRU_N);
    }

    // 3) layer-1 scan -> h1 (bf16), c1
    scan1_kernel<<<(SRU_B * 2 * SRU_N) / 256, 256, 0, stream>>>(
        u, hidden, mask, b1, h1b, hid);

    // 4) GEMM2: (8192 x 2048) x (2048 x 6144) -> u2 (reuses u buffer)
    {
        dim3 grid((unsigned)(LB / BM), (unsigned)(6 * SRU_N / BN));
        gemm_bf16_wmma<<<grid, 256, 0, stream>>>(h1b, w2t, u,
                                                 (int)LB, 6 * SRU_N, 2 * SRU_N);
    }

    // 5) layer-2 scan -> outputs + c2
    scan2_kernel<<<(SRU_B * 2 * SRU_N) / 256, 256, 0, stream>>>(
        u, hidden, mask, b2, h1b, outF, outB, hid);
}
